// GCN_Encoder_34342558499364
// MI455X (gfx1250) — matrix operand order repaired and checked by reading, not compiled, not run on hardware
//
#include <hip/hip_runtime.h>

// ---------------------------------------------------------------------------
// GCN encoder for MI455X (gfx1250, wave32).
//  - fp32 atomic scatter-add for graph aggregation (L2-resident, HBM-roofline
//    bound part); degree norms + BN affine fused into the gather.
//  - both 128x128 GEMMs per layer fused in one kernel using
//    v_wmma_f32_16x16x32_f16 (f16 in, f32 accumulate), weights pre-packed
//    into WMMA B-fragment layout. Accumulators start from inline-0 C matrices
//    (first K-tile peeled); biases folded into the epilogues.
// ---------------------------------------------------------------------------

typedef __attribute__((ext_vector_type(4)))  float    f4;
typedef __attribute__((ext_vector_type(8)))  float    v8f;
typedef __attribute__((ext_vector_type(8)))  _Float16 v8h;
typedef __attribute__((ext_vector_type(16))) _Float16 v16h;

#define HID 128
#define BN_EPS 1e-5f

// ---------------- degree kernels ----------------
__global__ void deg_kernel(const int* __restrict__ src, const int* __restrict__ dst,
                           int ne, float* __restrict__ dOut, float* __restrict__ dIn) {
    int e = blockIdx.x * blockDim.x + threadIdx.x;
    if (e >= ne) return;
    unsafeAtomicAdd(&dOut[src[e]], 1.0f);
    unsafeAtomicAdd(&dIn[dst[e]], 1.0f);
}

__global__ void rsq_kernel(float* __restrict__ a, long n) {
    long t = (long)blockIdx.x * blockDim.x + threadIdx.x;
    if (t >= n) return;
    a[t] = rsqrtf(fmaxf(a[t], 1.0f));
}

// ---------------- weight packing: f32 [K=128][N=128] -> f16 WMMA B-frags ----
// layout: [kt(4)][nt(8)][lane(32)][16 halves]; per lane: N = nt*16+(lane&15),
// halves j -> K = kt*32 + (lane>>4)*8 + j + (j>=8 ? 8 : 0)
__global__ void pack_w_kernel(const float* __restrict__ W, _Float16* __restrict__ out) {
    int t    = blockIdx.x * 256 + threadIdx.x;   // 0..1023
    int lane = t & 31;
    int nt   = (t >> 5) & 7;
    int kt   = t >> 8;
    int col  = nt * 16 + (lane & 15);
    int kb   = kt * 32 + (lane >> 4) * 8;
    _Float16* o = out + ((size_t)(kt * 8 + nt) * 32 + lane) * 16;
#pragma unroll
    for (int j = 0; j < 16; ++j)
        o[j] = (_Float16)W[(kb + j + (j >= 8 ? 8 : 0)) * HID + col];
}

// ---------------- fused scatter: agg[dst] += (a*u[src]+b) * rsqrt(outdeg) ---
// one wave per edge, 4 floats per lane, fp32 global atomics.
__global__ void scatter_kernel(const float* __restrict__ u, const float* __restrict__ ab,
                               int useAffine, const float* __restrict__ rsqOut,
                               const int* __restrict__ src, const int* __restrict__ dst,
                               float* __restrict__ agg, int ne) {
    long tid = (long)blockIdx.x * blockDim.x + threadIdx.x;
    int  e    = (int)(tid >> 5);
    int  lane = (int)(tid & 31);
    if (e >= ne) return;
    int   s  = src[e];
    int   d  = dst[e];
    float sc = rsqOut[s];
    f4 v = *(const f4*)(u + (size_t)s * HID + lane * 4);
    if (useAffine) {
        int c = lane * 4;
        f4 a = *(const f4*)(ab + c);
        f4 b = *(const f4*)(ab + HID + c);
        v = v * a + b;
    }
    v = v * sc;
    float* p = agg + (size_t)d * HID + lane * 4;
#pragma unroll
    for (int i = 0; i < 4; ++i) unsafeAtomicAdd(p + i, v[i]);
}

// ---------------- helpers for the fused GEMM --------------------------------
__device__ __forceinline__ v16h load_afrag(const float* __restrict__ pa, float scale) {
    // A-fragment (16-bit, 16x32): lanes 0-15 K={0..7,16..23}, lanes 16-31 +8
    f4 x0 = *(const f4*)(pa);
    f4 x1 = *(const f4*)(pa + 4);
    f4 x2 = *(const f4*)(pa + 16);
    f4 x3 = *(const f4*)(pa + 20);
    v16h af;
#pragma unroll
    for (int j = 0; j < 4; ++j) {
        af[j]      = (_Float16)(x0[j] * scale);
        af[4 + j]  = (_Float16)(x1[j] * scale);
        af[8 + j]  = (_Float16)(x2[j] * scale);
        af[12 + j] = (_Float16)(x3[j] * scale);
    }
    return af;
}

// ---------------- fused dual-GEMM per layer ---------------------------------
// Per wave: 16-row strip.  conv = sum_r (agg_r * rsqrt(indeg_r)) @ W_r + b_r
// then u = relu(conv @ W_fc + b_fc), write u, accumulate BN sum/sumsq.
__global__ void __launch_bounds__(128)
mega_gemm_kernel(const float* __restrict__ agg,    // 3 slabs of N*HID
                 const float* __restrict__ rsqIn,  // 3 slabs of N
                 int N,
                 const _Float16* __restrict__ wp,  // 4 packed matrices (seq,knn,dis,fc)
                 const float* __restrict__ bseq, const float* __restrict__ bknn,
                 const float* __restrict__ bdis, const float* __restrict__ bfc,
                 float* __restrict__ uOut, float* __restrict__ stats) {
    __shared__ _Float16 ldsConv[4][16][HID];
    __shared__ float    ssum[2 * HID];

    const int  tid  = threadIdx.x;
    const int  lane = tid & 31;
    const int  w    = tid >> 5;
    const int  mlo  = lane & 15;
    const int  hsel = lane >> 4;
    const int  kadd = hsel * 8;
    const long row0 = ((long)blockIdx.x * 4 + w) * 16;
    const bool active = (row0 + 16 <= N);

    ssum[tid]       = 0.0f;
    ssum[tid + 128] = 0.0f;
    __syncthreads();

    if (active) {
        const v16h* wpv   = (const v16h*)wp;
        const v8f   zeroC = {};

        // ---- GEMM 1: conv = sum_r scaled_agg_r @ W_r  (bias added at LDS stage)
        v8f c[8];
        {   // r = 0, kt = 0: peel so accumulators start from inline-0 C
            const float* A0 = agg + (row0 + mlo) * HID;
            const float  s0 = rsqIn[row0 + mlo];
            v16h af = load_afrag(A0 + kadd, s0);
            const v16h* bp = wpv + lane;
#pragma unroll
            for (int nt = 0; nt < 8; ++nt) {
                v16h bf = bp[nt * 32];
                c[nt] = __builtin_amdgcn_wmma_f32_16x16x32_f16(
                    false, af, false, bf, (short)0, zeroC, false, false);
            }
#pragma unroll
            for (int kt = 1; kt < 4; ++kt) {
                v16h af2 = load_afrag(A0 + kt * 32 + kadd, s0);
                const v16h* bp2 = wpv + (size_t)(kt * 8) * 32 + lane;
#pragma unroll
                for (int nt = 0; nt < 8; ++nt) {
                    v16h bf = bp2[nt * 32];
                    c[nt] = __builtin_amdgcn_wmma_f32_16x16x32_f16(
                        false, af2, false, bf, (short)0, c[nt], false, false);
                }
            }
        }
#pragma unroll
        for (int r = 1; r < 3; ++r) {
            const float* A     = agg + (size_t)r * N * HID + (row0 + mlo) * HID;
            const float  scale = rsqIn[(size_t)r * N + row0 + mlo];
#pragma unroll
            for (int kt = 0; kt < 4; ++kt) {
                v16h af = load_afrag(A + kt * 32 + kadd, scale);
                const v16h* bp = wpv + (size_t)(r * 32 + kt * 8) * 32 + lane;
#pragma unroll
                for (int nt = 0; nt < 8; ++nt) {
                    v16h bf = bp[nt * 32];
                    c[nt] = __builtin_amdgcn_wmma_f32_16x16x32_f16(
                        false, af, false, bf, (short)0, c[nt], false, false);
                }
            }
        }

        // ---- conv (+ summed relation biases) -> LDS (f16) for re-swizzle ----
        const int rbase = hsel * 8;
#pragma unroll
        for (int nt = 0; nt < 8; ++nt) {
            int   col = nt * 16 + mlo;
            float bs  = bseq[col] + bknn[col] + bdis[col];
#pragma unroll
            for (int i = 0; i < 8; ++i)
                ldsConv[w][rbase + i][col] = (_Float16)(c[nt][i] + bs);
        }
        // LDS ops from the same wave complete in order; wave-private region.

        // ---- GEMM 2: u = relu(conv @ W_fc + b_fc) ----
        v8f d[8];
        {   // kt = 0 peeled: inline-0 C
            const _Float16* pl = &ldsConv[w][mlo][kadd];
            v8h lo = *(const v8h*)(pl);
            v8h hi = *(const v8h*)(pl + 16);
            v16h af = __builtin_shufflevector(lo, hi, 0, 1, 2, 3, 4, 5, 6, 7,
                                              8, 9, 10, 11, 12, 13, 14, 15);
            const v16h* bp = wpv + (size_t)(3 * 32) * 32 + lane;
#pragma unroll
            for (int nt = 0; nt < 8; ++nt) {
                v16h bf = bp[nt * 32];
                d[nt] = __builtin_amdgcn_wmma_f32_16x16x32_f16(
                    false, af, false, bf, (short)0, zeroC, false, false);
            }
        }
#pragma unroll
        for (int kt = 1; kt < 4; ++kt) {
            const _Float16* pl = &ldsConv[w][mlo][kt * 32 + kadd];
            v8h lo = *(const v8h*)(pl);
            v8h hi = *(const v8h*)(pl + 16);
            v16h af = __builtin_shufflevector(lo, hi, 0, 1, 2, 3, 4, 5, 6, 7,
                                              8, 9, 10, 11, 12, 13, 14, 15);
            const v16h* bp = wpv + (size_t)(3 * 32 + kt * 8) * 32 + lane;
#pragma unroll
            for (int nt = 0; nt < 8; ++nt) {
                v16h bf = bp[nt * 32];
                d[nt] = __builtin_amdgcn_wmma_f32_16x16x32_f16(
                    false, af, false, bf, (short)0, d[nt], false, false);
            }
        }

        // ---- + b_fc, ReLU, store u, accumulate BN stats (LDS reduce) ----
        float* uo = uOut + row0 * HID;
#pragma unroll
        for (int nt = 0; nt < 8; ++nt) {
            int   col = nt * 16 + mlo;
            float bf  = bfc[col];
            float s = 0.0f, sq = 0.0f;
#pragma unroll
            for (int i = 0; i < 8; ++i) {
                float v = d[nt][i] + bf;
                v = v > 0.0f ? v : 0.0f;
                uo[(rbase + i) * HID + col] = v;
                s += v;
                sq += v * v;
            }
            atomicAdd(&ssum[col], s);
            atomicAdd(&ssum[HID + col], sq);
        }
    }

    __syncthreads();
    unsafeAtomicAdd(&stats[tid], ssum[tid]);
    unsafeAtomicAdd(&stats[tid + 128], ssum[tid + 128]);
}

// ---------------- BN: fold (gamma,beta,mean,var) into affine a,b ------------
__global__ void bn_finalize_kernel(const float* __restrict__ stats,
                                   const float* __restrict__ gamma,
                                   const float* __restrict__ beta,
                                   float invN, float* __restrict__ ab) {
    int   c    = threadIdx.x;  // 128
    float mean = stats[c] * invN;
    float var  = stats[HID + c] * invN - mean * mean;
    float a    = gamma[c] * rsqrtf(var + BN_EPS);
    ab[c]       = a;
    ab[HID + c] = beta[c] - mean * a;
}

// ---------------- final BN apply (only last layer materializes h) -----------
__global__ void bn_apply_kernel(const float* __restrict__ u, const float* __restrict__ ab,
                                float* __restrict__ out, long q) {
    long t = (long)blockIdx.x * blockDim.x + threadIdx.x;
    if (t >= q) return;
    f4  v = ((const f4*)u)[t];
    int c = (int)((t & 31) * 4);
    f4  a = *(const f4*)(ab + c);
    f4  b = *(const f4*)(ab + HID + c);
    ((f4*)out)[t] = v * a + b;
}

// ---------------------------------------------------------------------------
extern "C" void kernel_launch(void* const* d_in, const int* in_sizes, int n_in,
                              void* d_out, int out_size, void* d_ws, size_t ws_size,
                              hipStream_t stream) {
    (void)out_size; (void)ws_size;
    const int    N = in_sizes[0] / HID;
    const float* x = (const float*)d_in[0];

    // params: 3 layers x 10 tensors, dict keys flattened in sorted order:
    // 0:W_dis 1:W_fc 2:W_knn 3:W_seq 4:b_dis 5:b_fc 6:b_knn 7:b_seq 8:beta 9:gamma
    auto P = [&](int l, int k) { return (const float*)d_in[1 + l * 10 + k]; };

    const int  eb      = n_in - 6;  // src_seq,dst_seq,src_knn,dst_knn,src_dis,dst_dis
    const int* srcs[3] = {(const int*)d_in[eb + 0], (const int*)d_in[eb + 2], (const int*)d_in[eb + 4]};
    const int* dsts[3] = {(const int*)d_in[eb + 1], (const int*)d_in[eb + 3], (const int*)d_in[eb + 5]};
    const int  ne[3]   = {in_sizes[eb + 0], in_sizes[eb + 2], in_sizes[eb + 4]};

    // workspace carve-up
    char*  base = (char*)d_ws;
    size_t off  = 0;
    auto alloc = [&](size_t bytes) {
        char* p = base + off;
        off = (off + bytes + 255) & ~(size_t)255;
        return p;
    };
    float*    agg   = (float*)alloc((size_t)3 * N * HID * sizeof(float) + 2 * HID * sizeof(float));
    float*    stats = agg + (size_t)3 * N * HID;
    float*    uBuf  = (float*)alloc((size_t)N * HID * sizeof(float));
    float*    rsq   = (float*)alloc((size_t)6 * N * sizeof(float));  // out[3] then in[3]
    float*    ab    = (float*)alloc(2 * HID * sizeof(float));
    _Float16* wp    = (_Float16*)alloc((size_t)12 * 16384 * sizeof(_Float16));

    // --- degree norms (edges identical across layers: compute once) ---
    hipMemsetAsync(rsq, 0, (size_t)6 * N * sizeof(float), stream);
    for (int r = 0; r < 3; ++r)
        deg_kernel<<<(ne[r] + 255) / 256, 256, 0, stream>>>(
            srcs[r], dsts[r], ne[r], rsq + (size_t)r * N, rsq + (size_t)(3 + r) * N);
    {
        long tot = (long)6 * N;
        rsq_kernel<<<(unsigned)((tot + 255) / 256), 256, 0, stream>>>(rsq, tot);
    }

    // --- pack weights into f16 WMMA B-fragment layout (order: seq,knn,dis,fc)
    for (int l = 0; l < 3; ++l) {
        const int srcIdx[4] = {3, 2, 0, 1};  // W_seq, W_knn, W_dis, W_fc
        for (int m = 0; m < 4; ++m)
            pack_w_kernel<<<4, 256, 0, stream>>>(P(l, srcIdx[m]),
                                                 wp + ((size_t)l * 4 + m) * 16384);
    }

    // --- layers ---
    const float* uPrev = x;
    int aff = 0;
    for (int l = 0; l < 3; ++l) {
        hipMemsetAsync(agg, 0, (size_t)3 * N * HID * sizeof(float) + 2 * HID * sizeof(float), stream);
        for (int r = 0; r < 3; ++r) {
            long thr = (long)ne[r] * 32;
            scatter_kernel<<<(unsigned)((thr + 255) / 256), 256, 0, stream>>>(
                uPrev, ab, aff, rsq + (size_t)r * N, srcs[r], dsts[r],
                agg + (size_t)r * N * HID, ne[r]);
        }
        mega_gemm_kernel<<<(N + 63) / 64, 128, 0, stream>>>(
            agg, rsq + (size_t)3 * N, N, wp + (size_t)l * 4 * 16384,
            P(l, 7), P(l, 6), P(l, 4), P(l, 5), uBuf, stats);
        bn_finalize_kernel<<<1, 128, 0, stream>>>(stats, P(l, 9), P(l, 8),
                                                  1.0f / (float)N, ab);
        uPrev = uBuf;
        aff = 1;
    }
    long q = (long)N * (HID / 4);
    bn_apply_kernel<<<(unsigned)((q + 255) / 256), 256, 0, stream>>>(uBuf, ab, (float*)d_out, q);
}
